// ExpertRouter_58600533787171
// MI455X (gfx1250) — matrix operand order repaired and checked by reading, not compiled, not run on hardware
//
#include <hip/hip_runtime.h>

typedef unsigned int u32;
typedef unsigned short u16;
typedef __attribute__((ext_vector_type(4)))  u32   u32x4;
typedef __attribute__((ext_vector_type(8)))  u32   u32x8;
typedef __attribute__((ext_vector_type(4)))  u16   u16x4;
typedef __attribute__((ext_vector_type(16))) __bf16 v16bf;
typedef __attribute__((ext_vector_type(8)))  float v8f;

#define NTOK 32768   // B*T = 8*4096
#define DM   1024    // d_model
#define HD   512     // hidden
#define NE   64      // experts
#define LDSP 1032    // LDS row pitch in bf16: 516 dwords == 4 (mod 64) -> conflict-free

__device__ __forceinline__ u16 f32_to_bf16(float f) {
  u32 u = __float_as_uint(f);
  u32 r = (u + 0x7FFFu + ((u >> 16) & 1u)) >> 16;  // round-to-nearest-even
  return (u16)r;
}

__device__ __forceinline__ float gelu_exact(float v) {
  return 0.5f * v * (1.0f + erff(v * 0.70710678118654752440f));
}

union Frag { u32x8 u; v16bf b; };

// Load a 16-bf16 B fragment: two 16-byte slices per lane (K = hi*8..+7, 16+hi*8..+7).
__device__ __forceinline__ void load_bfrag(Frag& f, const u16* __restrict__ p) {
  u32x4 lo = *(const u32x4*)p;
  u32x4 hi = *(const u32x4*)(p + 16);
#pragma unroll
  for (int i = 0; i < 4; ++i) { f.u[i] = lo[i]; f.u[i + 4] = hi[i]; }
}

// Load a 16-bf16 A fragment from an LDS row: two 16-byte slices per lane.
__device__ __forceinline__ void load_afrag(Frag& f, const u16* aRow, int k0, int hi16) {
  u32x4 lo = *(const u32x4*)(aRow + k0 + hi16);        // K = hi*8 .. +7
  u32x4 hi = *(const u32x4*)(aRow + k0 + 16 + hi16);   // K = 16+hi*8 .. +7
#pragma unroll
  for (int i = 0; i < 4; ++i) { f.u[i] = lo[i]; f.u[i + 4] = hi[i]; }
}

// ---------------- weight pre-transpose + bf16 cast ----------------
__global__ __launch_bounds__(256) void k_convert_w1t(const float* __restrict__ W1,
                                                     u16* __restrict__ w1t) {
  int idx = blockIdx.x * 256 + threadIdx.x;        // over HD*DM
  int n = idx >> 10, k = idx & (DM - 1);
  w1t[idx] = f32_to_bf16(W1[(size_t)k * HD + n]);  // w1t[n][k] = W1[k][n]
}

__global__ __launch_bounds__(256) void k_convert_w2t(const float* __restrict__ W2,
                                                     u16* __restrict__ w2t) {
  int idx = blockIdx.x * 256 + threadIdx.x;        // over NE*HD
  int e = idx >> 9, k = idx & (HD - 1);
  w2t[idx] = f32_to_bf16(W2[(size_t)k * NE + e]);  // w2t[e][k] = W2[k][e]
}

// ---------- fused LayerNorm -> LDS (bf16) -> GEMM1 (WMMA) + GELU ----------
// One block = 128 tokens; activation tile lives in 264 KB of WGP LDS and is
// reused for all 8 N-chunks of W1. x is read from HBM exactly once.
__global__ __launch_bounds__(256) void k_ln_gemm1(const float* __restrict__ x,
                                                  const float* __restrict__ g,
                                                  const float* __restrict__ bb,
                                                  const u16* __restrict__ w1t,
                                                  const float* __restrict__ b1,
                                                  u16* __restrict__ h) {
  __shared__ u16 sA[128 * LDSP];                   // 264,192 B of 320 KB WGP LDS

  const int wave = threadIdx.x >> 5;
  const int lane = threadIdx.x & 31;
  const int laneLo = lane & 15;
  const int hiHalf = lane >> 4;
  const int tok0 = blockIdx.x * 128;

  // ---- Phase 1: LayerNorm, 16 tokens per wave, bf16 rows into LDS ----
  float4 gv[8], bv[8];
#pragma unroll
  for (int c = 0; c < 8; ++c) {
    gv[c] = *(const float4*)(g  + c * 128 + lane * 4);
    bv[c] = *(const float4*)(bb + c * 128 + lane * 4);
  }
  for (int t = 0; t < 16; ++t) {
    const int row = wave * 16 + t;
    const float* px = x + (size_t)(tok0 + row) * DM;
    float4 v[8];
    float s = 0.f, ss = 0.f;
#pragma unroll
    for (int c = 0; c < 8; ++c) {
      v[c] = *(const float4*)(px + c * 128 + lane * 4);
      s  += v[c].x + v[c].y + v[c].z + v[c].w;
      ss += v[c].x * v[c].x + v[c].y * v[c].y + v[c].z * v[c].z + v[c].w * v[c].w;
    }
#pragma unroll
    for (int off = 16; off > 0; off >>= 1) {
      s  += __shfl_xor(s,  off, 32);
      ss += __shfl_xor(ss, off, 32);
    }
    const float mu   = s * (1.0f / DM);
    const float rstd = rsqrtf(ss * (1.0f / DM) - mu * mu + 1e-5f);
    u16* pr = sA + row * LDSP;
#pragma unroll
    for (int c = 0; c < 8; ++c) {
      u16x4 o;
      o[0] = f32_to_bf16((v[c].x - mu) * rstd * gv[c].x + bv[c].x);
      o[1] = f32_to_bf16((v[c].y - mu) * rstd * gv[c].y + bv[c].y);
      o[2] = f32_to_bf16((v[c].z - mu) * rstd * gv[c].z + bv[c].z);
      o[3] = f32_to_bf16((v[c].w - mu) * rstd * gv[c].w + bv[c].w);
      *(u16x4*)(pr + c * 128 + lane * 4) = o;
    }
  }
  __syncthreads();

  // ---- Phase 2: wave-tile 16x64 GEMM, A from LDS, B double-buffered ----
  // K-loop fully unrolled: straight-line code -> no loop-carried phis, no
  // accumulator copies; loads sit one k-step ahead of their WMMAs.
  const u16* aRow = sA + (wave * 16 + laneLo) * LDSP;
  const int hi16 = hiHalf * 8;

#pragma unroll 1
  for (int n0 = 0; n0 < HD; n0 += 64) {
    v8f acc[4] = {};
    const u16* pb0 = w1t + (size_t)(n0 + laneLo) * DM + hiHalf * 8;
    Frag bufA[4], bufB[4];
#pragma unroll
    for (int j = 0; j < 4; ++j) load_bfrag(bufA[j], pb0 + (size_t)(j * 16) * DM);

#pragma unroll
    for (int i = 0; i < 16; ++i) {                 // 2 k-steps per iteration
      const int k0 = i * 64;
#pragma unroll
      for (int j = 0; j < 4; ++j)
        load_bfrag(bufB[j], pb0 + (size_t)(j * 16) * DM + (k0 + 32));
      Frag a0; load_afrag(a0, aRow, k0, hi16);
#pragma unroll
      for (int j = 0; j < 4; ++j)
        acc[j] = __builtin_amdgcn_wmma_f32_16x16x32_bf16(
            false, a0.b, false, bufA[j].b, (short)0, acc[j], false, false);
      if (i < 15) {                                // compile-time under unroll
#pragma unroll
        for (int j = 0; j < 4; ++j)
          load_bfrag(bufA[j], pb0 + (size_t)(j * 16) * DM + (k0 + 64));
      }
      Frag a1; load_afrag(a1, aRow, k0 + 32, hi16);
#pragma unroll
      for (int j = 0; j < 4; ++j)
        acc[j] = __builtin_amdgcn_wmma_f32_16x16x32_bf16(
            false, a1.b, false, bufB[j].b, (short)0, acc[j], false, false);
    }
#pragma unroll
    for (int j = 0; j < 4; ++j) {
      const int col = n0 + j * 16 + laneLo;
      const float bias = b1[col];
#pragma unroll
      for (int r = 0; r < 8; ++r) {
        const int row = tok0 + wave * 16 + hiHalf * 8 + r;  // C: M = r + hi*8
        h[(size_t)row * HD + col] = f32_to_bf16(gelu_exact(acc[j][r] + bias));
      }
    }
  }
}

// ---------------- GEMM2 (bf16 WMMA, double-buffered) + top-2 routing ----------------
__global__ __launch_bounds__(256) void k_gemm2_route(const u16* __restrict__ h,
                                                     const u16* __restrict__ w2t,
                                                     const float* __restrict__ b2,
                                                     float* __restrict__ out,
                                                     float* __restrict__ esum) {
  __shared__ float slog[128 * 64];
  __shared__ float ssum[64];
  __shared__ int   sI1[128], sI2[128];
  __shared__ float sP1[128], sP2[128];

  const int wave = threadIdx.x >> 5;
  const int lane = threadIdx.x & 31;
  const int laneLo = lane & 15;
  const int hiHalf = lane >> 4;
  const int hi16 = hiHalf * 8;
  const int m0 = wave * 16;
  const int tok0 = blockIdx.x * 128;

  if (threadIdx.x < 64) ssum[threadIdx.x] = 0.f;

  v8f acc[4] = {};
  const u16* pa  = h   + (size_t)(tok0 + m0 + laneLo) * HD;
  const u16* pb0 = w2t + (size_t)laneLo * HD + hiHalf * 8;

  Frag bufA[4], bufB[4];
#pragma unroll
  for (int j = 0; j < 4; ++j) load_bfrag(bufA[j], pb0 + (size_t)(j * 16) * HD);

#pragma unroll
  for (int i = 0; i < 8; ++i) {                    // 2 k-steps per iteration
    const int k0 = i * 64;
#pragma unroll
    for (int j = 0; j < 4; ++j)
      load_bfrag(bufB[j], pb0 + (size_t)(j * 16) * HD + (k0 + 32));
    Frag a0;
    {
      u32x4 lo = *(const u32x4*)(pa + k0 + hi16);
      u32x4 hi = *(const u32x4*)(pa + k0 + 16 + hi16);
#pragma unroll
      for (int q = 0; q < 4; ++q) { a0.u[q] = lo[q]; a0.u[q + 4] = hi[q]; }
    }
#pragma unroll
    for (int j = 0; j < 4; ++j)
      acc[j] = __builtin_amdgcn_wmma_f32_16x16x32_bf16(
          false, a0.b, false, bufA[j].b, (short)0, acc[j], false, false);
    if (i < 7) {                                   // compile-time under unroll
#pragma unroll
      for (int j = 0; j < 4; ++j)
        load_bfrag(bufA[j], pb0 + (size_t)(j * 16) * HD + (k0 + 64));
    }
    Frag a1;
    {
      u32x4 lo = *(const u32x4*)(pa + k0 + 32 + hi16);
      u32x4 hi = *(const u32x4*)(pa + k0 + 48 + hi16);
#pragma unroll
      for (int q = 0; q < 4; ++q) { a1.u[q] = lo[q]; a1.u[q + 4] = hi[q]; }
    }
#pragma unroll
    for (int j = 0; j < 4; ++j)
      acc[j] = __builtin_amdgcn_wmma_f32_16x16x32_bf16(
          false, a1.b, false, bufB[j].b, (short)0, acc[j], false, false);
  }
#pragma unroll
  for (int j = 0; j < 4; ++j) {
    const int col = j * 16 + laneLo;
    const float bias = b2[col];
#pragma unroll
    for (int r = 0; r < 8; ++r) {
      const int trow = m0 + hiHalf * 8 + r;
      slog[trow * 64 + col] = acc[j][r] + bias;    // TEMPERATURE == 1.0
    }
  }
  __syncthreads();

  if (threadIdx.x < 128) {
    const int t = threadIdx.x;
    float m1 = -3.4e38f, m2 = -3.4e38f;
    int i1 = 0, i2 = 0;
    for (int e = 0; e < 64; ++e) {
      const float l = slog[t * 64 + e];
      if (l > m1)      { m2 = m1; i2 = i1; m1 = l; i1 = e; }
      else if (l > m2) { m2 = l; i2 = e; }
    }
    const float p1 = 1.0f / (1.0f + expf(m2 - m1));  // softmax over top-2
    const float p2 = 1.0f - p1;
    sI1[t] = i1; sI2[t] = i2; sP1[t] = p1; sP2[t] = p2;
    atomicAdd(&ssum[i1], p1);
    atomicAdd(&ssum[i2], p2);
  }
  __syncthreads();

  for (int idx = threadIdx.x; idx < 128 * 64; idx += 256) {
    const int t = idx >> 6, e = idx & 63;
    const float v = (e == sI1[t]) ? sP1[t] : ((e == sI2[t]) ? sP2[t] : 0.0f);
    out[(size_t)(tok0 + t) * 64 + e] = v;
  }
  if (threadIdx.x < 64) atomicAdd(&esum[threadIdx.x], ssum[threadIdx.x]);
}

// ---------------- aux loss = sum_e (avg_e - 1/64)^2 ----------------
__global__ void k_aux(const float* __restrict__ esum, float* __restrict__ out) {
  const int lane = threadIdx.x;                    // 32 threads
  float a = 0.f;
  for (int e = lane; e < NE; e += 32) {
    const float d = esum[e] * (1.0f / (float)NTOK) - (1.0f / (float)NE);
    a += d * d;
  }
#pragma unroll
  for (int off = 16; off > 0; off >>= 1) a += __shfl_xor(a, off, 32);
  if (lane == 0) out[0] = a;
}

extern "C" void kernel_launch(void* const* d_in, const int* in_sizes, int n_in,
                              void* d_out, int out_size, void* d_ws, size_t ws_size,
                              hipStream_t stream) {
  const float* x    = (const float*)d_in[0];
  const float* ln_g = (const float*)d_in[1];
  const float* ln_b = (const float*)d_in[2];
  const float* W1   = (const float*)d_in[3];
  const float* b1   = (const float*)d_in[4];
  const float* W2   = (const float*)d_in[5];
  const float* b2   = (const float*)d_in[6];
  float* out = (float*)d_out;

  char* ws = (char*)d_ws;
  u16* w1t  = (u16*)ws;  ws += (size_t)HD * DM * sizeof(u16);    // 1 MB
  u16* w2t  = (u16*)ws;  ws += (size_t)NE * HD * sizeof(u16);    // 64 KB
  u16* hbuf = (u16*)ws;  ws += (size_t)NTOK * HD * sizeof(u16);  // 32 MB
  float* esum = (float*)ws;                                      // 256 B

  hipMemsetAsync(esum, 0, NE * sizeof(float), stream);
  k_convert_w1t<<<(HD * DM) / 256, 256, 0, stream>>>(W1, w1t);
  k_convert_w2t<<<(NE * HD) / 256, 256, 0, stream>>>(W2, w2t);
  k_ln_gemm1<<<NTOK / 128, 256, 0, stream>>>(x, ln_g, ln_b, w1t, b1, hbuf);
  k_gemm2_route<<<NTOK / 128, 256, 0, stream>>>(hbuf, w2t, b2, out, esum);
  k_aux<<<1, 32, 0, stream>>>(esum, out + (size_t)NTOK * NE);
}
